// Grok1MoE_2542620639595
// MI455X (gfx1250) — compile-verified, matrix operand order checked by reading, not run on hardware
//
#include <hip/hip_runtime.h>
#include <hip/hip_bf16.h>

typedef __attribute__((ext_vector_type(16))) __bf16 v16bf;
typedef __attribute__((ext_vector_type(8)))  __bf16 v8bf;
typedef __attribute__((ext_vector_type(4)))  __bf16 v4bf;
typedef __attribute__((ext_vector_type(2)))  __bf16 v2bf;
typedef __attribute__((ext_vector_type(8)))  float  v8f;

#define T_DIM 2048
#define D_DIM 2048
#define F_DIM 4096
#define E_NUM 8
#define K_TOP 2
#define SOFTCAP 30.0f

#define BM 128
#define BN 64
#define BK 32
#define LDSP 40   // padded row (bf16): 80B rows -> 16B aligned, conflict-free bank walk

// ---- workspace layout (bytes) ----
#define HB_OFF    0
#define ACT_OFF   (HB_OFF + T_DIM*D_DIM*2)
#define RMAP_OFF  (ACT_OFF + (size_t)T_DIM*K_TOP*F_DIM*2)
#define RW_OFF    (RMAP_OFF + T_DIM*K_TOP*4)
#define EID_OFF   (RW_OFF   + T_DIM*K_TOP*4)
#define TW_OFF    (EID_OFF  + T_DIM*K_TOP*4)
#define CNT_OFF   (TW_OFF   + T_DIM*K_TOP*4)
#define OFFS_OFF  (CNT_OFF  + 64)
#define CUR_OFF   (OFFS_OFF + 64)

// Build a v16bf WMMA fragment from an LDS row: elems [kb..kb+7] and [kb+16..kb+23]
__device__ __forceinline__ v16bf ld_frag(const __bf16* rowptr, int kb) {
    v8bf lo = *(const v8bf*)(rowptr + kb);
    v8bf hi = *(const v8bf*)(rowptr + kb + 16);
    return __builtin_shufflevector(lo, hi, 0,1,2,3,4,5,6,7,8,9,10,11,12,13,14,15);
}

__device__ __forceinline__ float gelu_tanh(float x) {
    float x3 = x * x * x;
    return 0.5f * x * (1.0f + tanhf(0.7978845608028654f * (x + 0.044715f * x3)));
}

// Pack two floats (k, k+1) into one bf16 pair, 4B LDS store (-> ds_store_b32/2addr).
__device__ __forceinline__ void pack_st(__bf16* dst, float lo, float hi) {
    v2bf p; p[0] = (__bf16)lo; p[1] = (__bf16)hi;
    *(v2bf*)dst = p;
}

// ---------------- small kernels ----------------
__global__ void k_init(int* counts, int* cursor) {
    int i = threadIdx.x;
    if (i < E_NUM) { counts[i] = 0; cursor[i] = 0; }
}

__global__ void k_zero_out(float* __restrict__ out) {
    size_t i = ((size_t)blockIdx.x * blockDim.x + threadIdx.x) * 4;
    *(float4*)(out + i) = make_float4(0.f, 0.f, 0.f, 0.f);
}

__global__ void k_cast(const float* __restrict__ H, __bf16* __restrict__ Hb) {
    size_t i = ((size_t)blockIdx.x * blockDim.x + threadIdx.x) * 4;
    float4 f = *(const float4*)(H + i);
    v4bf v; v[0] = (__bf16)f.x; v[1] = (__bf16)f.y; v[2] = (__bf16)f.z; v[3] = (__bf16)f.w;
    *(v4bf*)(Hb + i) = v;
}

__global__ __launch_bounds__(256)
void k_router(const float* __restrict__ H, const float* __restrict__ GW,
              int* __restrict__ eid, float* __restrict__ tw, int* __restrict__ counts) {
    const int t    = blockIdx.x;
    const int lane = threadIdx.x & 31;
    const int wave = threadIdx.x >> 5;
    const float* h = H + (size_t)t * D_DIM;
    float s = 0.f;
    for (int d = lane; d < D_DIM; d += 32) s += h[d] * GW[d * E_NUM + wave];
    #pragma unroll
    for (int off = 16; off > 0; off >>= 1) s += __shfl_down(s, off);
    __shared__ float lg[E_NUM];
    if (lane == 0) lg[wave] = s;
    __syncthreads();
    if (threadIdx.x == 0) {
        float v[E_NUM];
        #pragma unroll
        for (int e = 0; e < E_NUM; ++e) v[e] = tanhf(lg[e] / SOFTCAP) * SOFTCAP;
        int i0 = 0;
        for (int e = 1; e < E_NUM; ++e) if (v[e] > v[i0]) i0 = e;
        int i1 = (i0 == 0) ? 1 : 0;
        for (int e = 0; e < E_NUM; ++e) { if (e == i0) continue; if (v[e] > v[i1]) i1 = e; }
        float z  = expf(v[i1] - v[i0]);
        float w0 = 1.f / (1.f + z);
        float w1 = z   / (1.f + z);
        eid[t*K_TOP]   = i0;  eid[t*K_TOP+1] = i1;
        tw [t*K_TOP]   = w0;  tw [t*K_TOP+1] = w1;
        atomicAdd(&counts[i0], 1);
        atomicAdd(&counts[i1], 1);
    }
}

__global__ void k_offsets(const int* __restrict__ counts, int* __restrict__ offs,
                          int* __restrict__ cursor) {
    if (threadIdx.x == 0) {
        int run = 0;
        for (int e = 0; e < E_NUM; ++e) { offs[e] = run; run += counts[e]; cursor[e] = 0; }
    }
}

__global__ void k_scatter(const int* __restrict__ eid, const float* __restrict__ tw,
                          const int* __restrict__ offs, int* __restrict__ cursor,
                          int* __restrict__ rowmap, float* __restrict__ roww) {
    int s = blockIdx.x * blockDim.x + threadIdx.x;
    if (s >= T_DIM * K_TOP) return;
    int e   = eid[s];
    int pos = atomicAdd(&cursor[e], 1);
    rowmap[offs[e] + pos] = s >> 1;
    roww  [offs[e] + pos] = tw[s];
}

// ---------------- fused gate/up GEMM + gelu epilogue ----------------
// grid: (T/BM fastest, F/BN, E). Double-buffered LDS, 1 barrier/iter:
//   wait+store(it) -> issue loads(it+1) -> barrier -> WMMA(it)
__global__ __launch_bounds__(256)
void k_gateup(const __bf16* __restrict__ Hb, const float* __restrict__ Wg,
              const float* __restrict__ Wu, __bf16* __restrict__ act,
              const int* __restrict__ rowmap, const int* __restrict__ offs,
              const int* __restrict__ counts) {
    const int e   = blockIdx.z;
    const int cnt = counts[e];
    const int row0 = blockIdx.x * BM;
    if (row0 >= cnt) return;
    const int fn   = blockIdx.y * BN;
    const int base = offs[e];

    __shared__ alignas(16) __bf16 As[2][BM][LDSP];
    __shared__ alignas(16) __bf16 Bg[2][BN][LDSP];   // [n][k], K-contiguous
    __shared__ alignas(16) __bf16 Bu[2][BN][LDSP];

    const int tid  = threadIdx.x;
    const int lane = tid & 31;
    const int wave = tid >> 5;
    const int wn    = wave & 3;
    const int whalf = wave >> 2;
    const int half = lane >> 4;
    const int idx  = lane & 15;
    const int kb   = half * 8;

    // A-fill: 2 threads per row, 16 bf16 each
    const int am = tid >> 1;
    const int aj = (tid & 1) * 16;
    int tok = 0;
    if (row0 + am < cnt) tok = rowmap[base + row0 + am];
    const __bf16* ap = Hb + (size_t)tok * D_DIM + aj;

    // B-fill: thread covers n0..n0+3 x {k, k+1}
    const int n0 = (tid & 15) * 4;
    const int kk = (tid >> 4) * 2;
    const float* pg = Wg + (size_t)e * D_DIM * F_DIM + (size_t)kk * F_DIM + fn + n0;
    const float* pu = Wu + (size_t)e * D_DIM * F_DIM + (size_t)kk * F_DIM + fn + n0;

    v8f accg[4] = {{0}, {0}, {0}, {0}};
    v8f accu[4] = {{0}, {0}, {0}, {0}};

    const int niter = D_DIM / BK;
    uint4 sa0, sa1; float4 g0, g1, u0, u1;

    // prologue: stage iter 0
    sa0 = *(const uint4*)(ap);
    sa1 = *(const uint4*)(ap + 8);
    g0 = *(const float4*)(pg);  g1 = *(const float4*)(pg + F_DIM);
    u0 = *(const float4*)(pu);  u1 = *(const float4*)(pu + F_DIM);
    ap += BK;
    pg += (size_t)BK * F_DIM;
    pu += (size_t)BK * F_DIM;

    for (int it = 0; it < niter; ++it) {
        const int cur = it & 1;
        // commit staged tile (waits only on its own loads)
        *(uint4*)&As[cur][am][aj]     = sa0;
        *(uint4*)&As[cur][am][aj + 8] = sa1;
        pack_st(&Bg[cur][n0+0][kk], g0.x, g1.x);  pack_st(&Bg[cur][n0+1][kk], g0.y, g1.y);
        pack_st(&Bg[cur][n0+2][kk], g0.z, g1.z);  pack_st(&Bg[cur][n0+3][kk], g0.w, g1.w);
        pack_st(&Bu[cur][n0+0][kk], u0.x, u1.x);  pack_st(&Bu[cur][n0+1][kk], u0.y, u1.y);
        pack_st(&Bu[cur][n0+2][kk], u0.z, u1.z);  pack_st(&Bu[cur][n0+3][kk], u0.w, u1.w);

        // prefetch next tile: in flight across the barrier and all WMMAs below
        if (it + 1 < niter) {
            sa0 = *(const uint4*)(ap);
            sa1 = *(const uint4*)(ap + 8);
            g0 = *(const float4*)(pg);  g1 = *(const float4*)(pg + F_DIM);
            u0 = *(const float4*)(pu);  u1 = *(const float4*)(pu + F_DIM);
            ap += BK;
            pg += (size_t)BK * F_DIM;
            pu += (size_t)BK * F_DIM;
        }

        __syncthreads();

        v16bf bg = ld_frag(&Bg[cur][wn*16 + idx][0], kb);
        v16bf bu = ld_frag(&Bu[cur][wn*16 + idx][0], kb);
        #pragma unroll
        for (int i = 0; i < 4; ++i) {
            v16bf a = ld_frag(&As[cur][whalf*64 + i*16 + idx][0], kb);
            accg[i] = __builtin_amdgcn_wmma_f32_16x16x32_bf16(false, a, false, bg, (short)0, accg[i], false, false);
            accu[i] = __builtin_amdgcn_wmma_f32_16x16x32_bf16(false, a, false, bu, (short)0, accu[i], false, false);
        }
        // no trailing barrier: next iter writes the other buffer; a wave can only
        // be 1 buffer ahead because of the single barrier above.
    }

    const int n = fn + wn * 16 + idx;
    #pragma unroll
    for (int i = 0; i < 4; ++i) {
        #pragma unroll
        for (int r = 0; r < 8; ++r) {
            int grow = row0 + whalf*64 + i*16 + half*8 + r;
            if (grow < cnt) {
                float v = gelu_tanh(accg[i][r]) * accu[i][r];
                act[(size_t)(base + grow) * F_DIM + n] = (__bf16)v;
            }
        }
    }
}

// ---------------- down GEMM + weighted scatter-add epilogue ----------------
// grid: (T/BM fastest, D/BN, E). Same double-buffered pipeline.
__global__ __launch_bounds__(256)
void k_down(const __bf16* __restrict__ act, const float* __restrict__ Wd,
            float* __restrict__ out, const int* __restrict__ rowmap,
            const float* __restrict__ roww, const int* __restrict__ offs,
            const int* __restrict__ counts) {
    const int e   = blockIdx.z;
    const int cnt = counts[e];
    const int row0 = blockIdx.x * BM;
    if (row0 >= cnt) return;
    const int dn   = blockIdx.y * BN;
    const int base = offs[e];

    __shared__ alignas(16) __bf16 As[2][BM][LDSP];
    __shared__ alignas(16) __bf16 Bd[2][BN][LDSP];

    const int tid  = threadIdx.x;
    const int lane = tid & 31;
    const int wave = tid >> 5;
    const int wn    = wave & 3;
    const int whalf = wave >> 2;
    const int half = lane >> 4;
    const int idx  = lane & 15;
    const int kb   = half * 8;

    const int am = tid >> 1;
    const int aj = (tid & 1) * 16;
    const bool avalid = (row0 + am < cnt);
    const __bf16* ap = act + (size_t)(base + row0 + am) * F_DIM + aj;

    const int n0 = (tid & 15) * 4;
    const int kk = (tid >> 4) * 2;
    const float* pb = Wd + (size_t)e * F_DIM * D_DIM + (size_t)kk * D_DIM + dn + n0;

    v8f acc[4] = {{0}, {0}, {0}, {0}};

    const int niter = F_DIM / BK;
    uint4 sa0 = make_uint4(0,0,0,0), sa1 = make_uint4(0,0,0,0);
    float4 b0, b1;

    if (avalid) { sa0 = *(const uint4*)(ap); sa1 = *(const uint4*)(ap + 8); }
    b0 = *(const float4*)(pb);  b1 = *(const float4*)(pb + D_DIM);
    ap += BK;
    pb += (size_t)BK * D_DIM;

    for (int it = 0; it < niter; ++it) {
        const int cur = it & 1;
        *(uint4*)&As[cur][am][aj]     = sa0;
        *(uint4*)&As[cur][am][aj + 8] = sa1;
        pack_st(&Bd[cur][n0+0][kk], b0.x, b1.x);  pack_st(&Bd[cur][n0+1][kk], b0.y, b1.y);
        pack_st(&Bd[cur][n0+2][kk], b0.z, b1.z);  pack_st(&Bd[cur][n0+3][kk], b0.w, b1.w);

        if (it + 1 < niter) {
            if (avalid) {
                sa0 = *(const uint4*)(ap);
                sa1 = *(const uint4*)(ap + 8);
            }
            b0 = *(const float4*)(pb);  b1 = *(const float4*)(pb + D_DIM);
            ap += BK;
            pb += (size_t)BK * D_DIM;
        }

        __syncthreads();

        v16bf b = ld_frag(&Bd[cur][wn*16 + idx][0], kb);
        #pragma unroll
        for (int i = 0; i < 4; ++i) {
            v16bf a = ld_frag(&As[cur][whalf*64 + i*16 + idx][0], kb);
            acc[i] = __builtin_amdgcn_wmma_f32_16x16x32_bf16(false, a, false, b, (short)0, acc[i], false, false);
        }
    }

    const int n = dn + wn * 16 + idx;
    #pragma unroll
    for (int i = 0; i < 4; ++i) {
        #pragma unroll
        for (int r = 0; r < 8; ++r) {
            int grow = row0 + whalf*64 + i*16 + half*8 + r;
            if (grow < cnt) {
                int   tok = rowmap[base + grow];
                float w   = roww [base + grow];
                atomicAdd(&out[(size_t)tok * D_DIM + n], w * acc[i][r]);
            }
        }
    }
}

// ---------------- host launcher ----------------
extern "C" void kernel_launch(void* const* d_in, const int* in_sizes, int n_in,
                              void* d_out, int out_size, void* d_ws, size_t ws_size,
                              hipStream_t stream) {
    const float* H  = (const float*)d_in[0];   // [T, D]
    const float* GW = (const float*)d_in[1];   // [D, E]
    const float* Wg = (const float*)d_in[2];   // [E, D, F]
    const float* Wu = (const float*)d_in[3];   // [E, D, F]
    const float* Wd = (const float*)d_in[4];   // [E, F, D]
    float* out = (float*)d_out;                // [T, D]

    char* ws = (char*)d_ws;
    __bf16* Hb     = (__bf16*)(ws + HB_OFF);
    __bf16* act    = (__bf16*)(ws + ACT_OFF);
    int*    rowmap = (int*)   (ws + RMAP_OFF);
    float*  roww   = (float*) (ws + RW_OFF);
    int*    eid    = (int*)   (ws + EID_OFF);
    float*  tw     = (float*) (ws + TW_OFF);
    int*    counts = (int*)   (ws + CNT_OFF);
    int*    offs   = (int*)   (ws + OFFS_OFF);
    int*    cursor = (int*)   (ws + CUR_OFF);

    k_init<<<1, 32, 0, stream>>>(counts, cursor);
    k_zero_out<<<(T_DIM * D_DIM) / (256 * 4), 256, 0, stream>>>(out);
    k_cast<<<(T_DIM * D_DIM) / (256 * 4), 256, 0, stream>>>(H, Hb);
    k_router<<<T_DIM, 256, 0, stream>>>(H, GW, eid, tw, counts);
    k_offsets<<<1, 1, 0, stream>>>(counts, offs, cursor);
    k_scatter<<<(T_DIM * K_TOP) / 256, 256, 0, stream>>>(eid, tw, offs, cursor, rowmap, roww);

    dim3 g1(T_DIM / BM, F_DIM / BN, E_NUM);
    k_gateup<<<g1, 256, 0, stream>>>(Hb, Wg, Wu, act, rowmap, offs, counts);

    dim3 g2(T_DIM / BM, D_DIM / BN, E_NUM);
    k_down<<<g2, 256, 0, stream>>>(act, Wd, out, rowmap, roww, offs, counts);
}
